// GCN_54236847014390
// MI455X (gfx1250) — matrix operand order repaired and checked by reading, not compiled, not run on hardware
//
#include <hip/hip_runtime.h>
#include <hip/hip_bf16.h>

// ---------------------------------------------------------------------------
// 2-layer GAT (PyG GATConv semantics) for MI455X / gfx1250.
//  - GEMMs run on v_wmma_f32_16x16x32_f16 (f16 A/B, f32 accum)
//  - attention coefficients folded into the GEMM as extra output columns
//  - segment softmax via order-preserving-uint atomicMax + f32 atomicAdd
//  - wave-per-edge coalesced scatter-add aggregation
// ---------------------------------------------------------------------------

typedef __attribute__((ext_vector_type(16))) _Float16 v16h;
typedef __attribute__((ext_vector_type(8)))  _Float16 v8h;
typedef __attribute__((ext_vector_type(8)))  float    v8f;

#define ENC_NEG_INF 0x007FFFFFu   // encode(-inf)

__device__ __forceinline__ unsigned enc_f32(float f) {
    unsigned u = __float_as_uint(f);
    return (u & 0x80000000u) ? ~u : (u | 0x80000000u);
}
__device__ __forceinline__ float dec_f32(unsigned u) {
    return __uint_as_float((u & 0x80000000u) ? (u & 0x7fffffffu) : ~u);
}
__device__ __forceinline__ float lrelu(float v) { return v > 0.f ? v : 0.2f * v; }
__device__ __forceinline__ float elu(float v)   { return v > 0.f ? v : (expf(v) - 1.f); }

// ---------------------------------------------------------------------------
// Prep: Wt1ext[col][k], col 0..127 = W1^T, 128..131 = W1.a_src (per head),
// 132..135 = W1.a_dst, 136..143 = zero pad.  All f16, K-contiguous so the
// WMMA B-operand load is a single 32B contiguous load per lane.
// ---------------------------------------------------------------------------
__global__ void gat_prep_w1(const float* __restrict__ W1,
                            const float* __restrict__ asr,
                            const float* __restrict__ adr,
                            _Float16* __restrict__ wt) {
    int tid = blockIdx.x * blockDim.x + threadIdx.x;
    if (tid >= 144 * 128) return;
    int col = tid / 128, k = tid % 128;
    float v = 0.f;
    if (col < 128) {
        v = W1[k * 128 + col];
    } else if (col < 132) {
        int h = col - 128; float s = 0.f;
        for (int c = 0; c < 32; ++c) s += W1[k * 128 + h * 32 + c] * asr[h * 32 + c];
        v = s;
    } else if (col < 136) {
        int h = col - 132; float s = 0.f;
        for (int c = 0; c < 32; ++c) s += W1[k * 128 + h * 32 + c] * adr[h * 32 + c];
        v = s;
    }
    wt[col * 128 + k] = (_Float16)v;
}

// Wt2ext[col][k]: col 0..7 = W2^T, 8 = W2.a_src2, 9 = W2.a_dst2, 10..15 pad.
__global__ void gat_prep_w2(const float* __restrict__ W2,
                            const float* __restrict__ asr,
                            const float* __restrict__ adr,
                            _Float16* __restrict__ wt) {
    int tid = blockIdx.x * blockDim.x + threadIdx.x;
    if (tid >= 16 * 128) return;
    int col = tid / 128, k = tid % 128;
    float v = 0.f;
    if (col < 8) v = W2[k * 8 + col];
    else if (col == 8) { float s = 0.f; for (int c = 0; c < 8; ++c) s += W2[k * 8 + c] * asr[c]; v = s; }
    else if (col == 9) { float s = 0.f; for (int c = 0; c < 8; ++c) s += W2[k * 8 + c] * adr[c]; v = s; }
    wt[col * 128 + k] = (_Float16)v;
}

__global__ void gat_conv_f16(const float* __restrict__ src, _Float16* __restrict__ dst, int n) {
    int i = blockIdx.x * blockDim.x + threadIdx.x;
    if (i < n) dst[i] = (_Float16)src[i];
}

// Init softmax state + bias-preloaded accumulators (out += bias after segsum)
__global__ void gat_init(float* __restrict__ out1, const float* __restrict__ b1,
                         float* __restrict__ out2, const float* __restrict__ b2,
                         unsigned* __restrict__ m1u, float* __restrict__ den1,
                         unsigned* __restrict__ m2u, float* __restrict__ den2, int n) {
    int i = blockIdx.x * blockDim.x + threadIdx.x;
    if (i < n * 128) out1[i] = b1[i & 127];
    if (i < n * 8)   out2[i] = b2[i & 7];
    if (i < n * 4) { m1u[i] = ENC_NEG_INF; den1[i] = 0.f; }
    if (i < n)     { m2u[i] = ENC_NEG_INF; den2[i] = 0.f; }
}

// ---------------------------------------------------------------------------
// GEMM1: [16 rows] x [128 K] x [144 cols], 9 waves/block, 4 WMMA per wave.
// Waves 0..7 emit h1[N,128]; wave 8 emits alpha_src[N,4] / alpha_dst[N,4].
// ---------------------------------------------------------------------------
__global__ void gat_gemm1(const _Float16* __restrict__ xh,
                          const _Float16* __restrict__ wt,
                          float* __restrict__ h1,
                          float* __restrict__ as1, float* __restrict__ ad1, int n) {
    int wave = threadIdx.x >> 5;
    int lane = threadIdx.x & 31;
    int rowbase = blockIdx.x * 16;
    if (rowbase >= n) return;
    int colbase = wave * 16;
    int arow = rowbase + (lane & 15); if (arow >= n) arow = n - 1;
    int akoff = (lane < 16) ? 0 : 16;    // K-half split for A operand
    int bkoff = (lane < 16) ? 0 : 16;    // K-half split for B operand
    int bcol  = colbase + (lane & 15);

    v8f c = {};
    #pragma unroll
    for (int kk = 0; kk < 128; kk += 32) {
        v8h alo = *(const v8h*)(xh + (size_t)arow * 128 + kk + akoff);
        v8h ahi = *(const v8h*)(xh + (size_t)arow * 128 + kk + 8 + akoff);
        v16h a = __builtin_shufflevector(alo, ahi, 0,1,2,3,4,5,6,7,8,9,10,11,12,13,14,15);
        v16h b = *(const v16h*)(wt + (size_t)bcol * 128 + kk + bkoff);
        c = __builtin_amdgcn_wmma_f32_16x16x32_f16(false, a, false, b, (short)0, c, false, false);
    }

    int rbase = rowbase + ((lane < 16) ? 0 : 8);
    int cl = lane & 15;
    if (wave < 8) {
        #pragma unroll
        for (int r = 0; r < 8; ++r) {
            int nrow = rbase + r;
            if (nrow < n) h1[(size_t)nrow * 128 + colbase + cl] = c[r];
        }
    } else {
        #pragma unroll
        for (int r = 0; r < 8; ++r) {
            int nrow = rbase + r;
            if (nrow >= n) continue;
            if (cl < 4)      as1[nrow * 4 + cl] = c[r];
            else if (cl < 8) ad1[nrow * 4 + (cl - 4)] = c[r];
        }
    }
}

// GEMM2: one wave per 16 rows, K=128, 16 output cols (8 = h2, col8/9 = alphas)
__global__ void gat_gemm2(const _Float16* __restrict__ zh,
                          const _Float16* __restrict__ wt,
                          float* __restrict__ h2,
                          float* __restrict__ as2, float* __restrict__ ad2, int n) {
    int wave = threadIdx.x >> 5;
    int lane = threadIdx.x & 31;
    int rowbase = (blockIdx.x * 8 + wave) * 16;
    if (rowbase >= n) return;
    int arow = rowbase + (lane & 15); if (arow >= n) arow = n - 1;
    int akoff = (lane < 16) ? 0 : 16;
    int bkoff = (lane < 16) ? 0 : 16;
    int bcol  = lane & 15;

    v8f c = {};
    #pragma unroll
    for (int kk = 0; kk < 128; kk += 32) {
        v8h alo = *(const v8h*)(zh + (size_t)arow * 128 + kk + akoff);
        v8h ahi = *(const v8h*)(zh + (size_t)arow * 128 + kk + 8 + akoff);
        v16h a = __builtin_shufflevector(alo, ahi, 0,1,2,3,4,5,6,7,8,9,10,11,12,13,14,15);
        v16h b = *(const v16h*)(wt + (size_t)bcol * 128 + kk + bkoff);
        c = __builtin_amdgcn_wmma_f32_16x16x32_f16(false, a, false, b, (short)0, c, false, false);
    }

    int rbase = rowbase + ((lane < 16) ? 0 : 8);
    int cl = lane & 15;
    #pragma unroll
    for (int r = 0; r < 8; ++r) {
        int nrow = rbase + r;
        if (nrow >= n) continue;
        if (cl < 8)       h2[nrow * 8 + cl] = c[r];
        else if (cl == 8) as2[nrow] = c[r];
        else if (cl == 9) ad2[nrow] = c[r];
    }
}

// ---------------------------------------------------------------------------
// Edge passes, layer 1 (H=4).  idx<E: real edge; else self-loop (idx-E).
// ---------------------------------------------------------------------------
__global__ void gat_edge1a(const int* __restrict__ ei,
                           const float4* __restrict__ as1, const float4* __restrict__ ad1,
                           float4* __restrict__ ebuf, unsigned* __restrict__ m1u,
                           int E, int NE2) {
    int g = blockIdx.x * blockDim.x + threadIdx.x;
    if (g >= NE2) return;
    int s = (g < E) ? ei[g]     : (g - E);
    int d = (g < E) ? ei[E + g] : (g - E);
    float4 a = as1[s], b = ad1[d];
    float4 e;
    e.x = lrelu(a.x + b.x); e.y = lrelu(a.y + b.y);
    e.z = lrelu(a.z + b.z); e.w = lrelu(a.w + b.w);
    ebuf[g] = e;
    atomicMax(&m1u[d * 4 + 0], enc_f32(e.x));
    atomicMax(&m1u[d * 4 + 1], enc_f32(e.y));
    atomicMax(&m1u[d * 4 + 2], enc_f32(e.z));
    atomicMax(&m1u[d * 4 + 3], enc_f32(e.w));
}

__global__ void gat_edge1b(const int* __restrict__ ei,
                           float4* __restrict__ ebuf, const unsigned* __restrict__ m1u,
                           float* __restrict__ den1, int E, int NE2) {
    int g = blockIdx.x * blockDim.x + threadIdx.x;
    if (g >= NE2) return;
    int d = (g < E) ? ei[E + g] : (g - E);
    float4 e = ebuf[g];
    float4 ex;
    ex.x = expf(e.x - dec_f32(m1u[d * 4 + 0]));
    ex.y = expf(e.y - dec_f32(m1u[d * 4 + 1]));
    ex.z = expf(e.z - dec_f32(m1u[d * 4 + 2]));
    ex.w = expf(e.w - dec_f32(m1u[d * 4 + 3]));
    ebuf[g] = ex;
    atomicAdd(&den1[d * 4 + 0], ex.x);
    atomicAdd(&den1[d * 4 + 1], ex.y);
    atomicAdd(&den1[d * 4 + 2], ex.z);
    atomicAdd(&den1[d * 4 + 3], ex.w);
}

// One wave per edge: lane j-th chunk col = lane + 32*j belongs to head j.
__global__ void gat_edge1c(const int* __restrict__ ei,
                           const float4* __restrict__ ebuf, const float4* __restrict__ den1,
                           const float* __restrict__ h1, float* __restrict__ out1,
                           int E, int NE2) {
    int lane = threadIdx.x & 31;
    int g = blockIdx.x * 8 + (threadIdx.x >> 5);
    if (g >= NE2) return;
    int s = (g < E) ? ei[g]     : (g - E);
    int d = (g < E) ? ei[E + g] : (g - E);
    float4 ex = ebuf[g], dn = den1[d];
    float al[4];
    al[0] = ex.x / (dn.x + 1e-16f); al[1] = ex.y / (dn.y + 1e-16f);
    al[2] = ex.z / (dn.z + 1e-16f); al[3] = ex.w / (dn.w + 1e-16f);
    const float* hs = h1 + (size_t)s * 128;
    float* od = out1 + (size_t)d * 128;
    #pragma unroll
    for (int j = 0; j < 4; ++j) {
        int c = lane + 32 * j;
        atomicAdd(&od[c], hs[c] * al[j]);
    }
}

__global__ void gat_elu1(const float* __restrict__ out1, _Float16* __restrict__ zh, int n) {
    int i = blockIdx.x * blockDim.x + threadIdx.x;
    if (i < n) zh[i] = (_Float16)elu(out1[i]);
}

// ---------------------------------------------------------------------------
// Edge passes, layer 2 (H=1, C=8)
// ---------------------------------------------------------------------------
__global__ void gat_edge2a(const int* __restrict__ ei,
                           const float* __restrict__ as2, const float* __restrict__ ad2,
                           float* __restrict__ ebuf, unsigned* __restrict__ m2u,
                           int E, int NE2) {
    int g = blockIdx.x * blockDim.x + threadIdx.x;
    if (g >= NE2) return;
    int s = (g < E) ? ei[g]     : (g - E);
    int d = (g < E) ? ei[E + g] : (g - E);
    float e = lrelu(as2[s] + ad2[d]);
    ebuf[g] = e;
    atomicMax(&m2u[d], enc_f32(e));
}

__global__ void gat_edge2b(const int* __restrict__ ei,
                           float* __restrict__ ebuf, const unsigned* __restrict__ m2u,
                           float* __restrict__ den2, int E, int NE2) {
    int g = blockIdx.x * blockDim.x + threadIdx.x;
    if (g >= NE2) return;
    int d = (g < E) ? ei[E + g] : (g - E);
    float ex = expf(ebuf[g] - dec_f32(m2u[d]));
    ebuf[g] = ex;
    atomicAdd(&den2[d], ex);
}

// 8 threads per edge (one per output channel)
__global__ void gat_edge2c(const int* __restrict__ ei,
                           const float* __restrict__ ebuf, const float* __restrict__ den2,
                           const float* __restrict__ h2, float* __restrict__ out2,
                           int E, int NE2) {
    int gid = blockIdx.x * blockDim.x + threadIdx.x;
    int g = gid >> 3, c = gid & 7;
    if (g >= NE2) return;
    int s = (g < E) ? ei[g]     : (g - E);
    int d = (g < E) ? ei[E + g] : (g - E);
    float alpha = ebuf[g] / (den2[d] + 1e-16f);
    atomicAdd(&out2[d * 8 + c], h2[s * 8 + c] * alpha);
}

__global__ void gat_final(const float* __restrict__ out2,
                          const float* __restrict__ Wlin, const float* __restrict__ blin,
                          float* __restrict__ out, int n) {
    int i = blockIdx.x * blockDim.x + threadIdx.x;
    if (i >= n) return;
    float acc = blin[0];
    #pragma unroll
    for (int c = 0; c < 8; ++c) acc += elu(out2[i * 8 + c]) * Wlin[c];
    out[i] = 1.f / (1.f + expf(-acc));
}

// ---------------------------------------------------------------------------
extern "C" void kernel_launch(void* const* d_in, const int* in_sizes, int n_in,
                              void* d_out, int out_size, void* d_ws, size_t ws_size,
                              hipStream_t stream) {
    const float* x    = (const float*)d_in[0];
    const int*   ei   = (const int*)d_in[1];
    const float* W1   = (const float*)d_in[3];
    const float* a_s1 = (const float*)d_in[4];
    const float* a_d1 = (const float*)d_in[5];
    const float* b1   = (const float*)d_in[6];
    const float* W2   = (const float*)d_in[7];
    const float* a_s2 = (const float*)d_in[8];
    const float* a_d2 = (const float*)d_in[9];
    const float* b2   = (const float*)d_in[10];
    const float* Wlin = (const float*)d_in[11];
    const float* blin = (const float*)d_in[12];
    float* out = (float*)d_out;

    const int N   = in_sizes[0] / 128;
    const int E   = in_sizes[1] / 2;
    const int NE2 = E + N;

    char* ws = (char*)d_ws;
    size_t off = 0;
    auto carve = [&](size_t bytes) -> void* {
        void* p = (void*)(ws + off);
        off = (off + bytes + 255) & ~(size_t)255;
        return p;
    };

    _Float16* xh   = (_Float16*)carve((size_t)N * 128 * 2);
    _Float16* wt1  = (_Float16*)carve(144 * 128 * 2);
    _Float16* wt2  = (_Float16*)carve(16 * 128 * 2);
    float*    h1   = (float*)carve((size_t)N * 128 * 4);
    float*    as1  = (float*)carve((size_t)N * 4 * 4);
    float*    ad1  = (float*)carve((size_t)N * 4 * 4);
    unsigned* m1u  = (unsigned*)carve((size_t)N * 4 * 4);
    float*    den1 = (float*)carve((size_t)N * 4 * 4);
    float*    eb1  = (float*)carve((size_t)NE2 * 4 * 4);
    float*    out1 = (float*)carve((size_t)N * 128 * 4);
    _Float16* z1h  = (_Float16*)carve((size_t)N * 128 * 2);
    float*    h2   = (float*)carve((size_t)N * 8 * 4);
    float*    as2  = (float*)carve((size_t)N * 4);
    float*    ad2  = (float*)carve((size_t)N * 4);
    unsigned* m2u  = (unsigned*)carve((size_t)N * 4);
    float*    den2 = (float*)carve((size_t)N * 4);
    float*    eb2  = (float*)carve((size_t)NE2 * 4);
    float*    out2 = (float*)carve((size_t)N * 8 * 4);

    const int T = 256;
    // weight prep + input conversion + state init
    gat_prep_w1<<<(144 * 128 + T - 1) / T, T, 0, stream>>>(W1, a_s1, a_d1, wt1);
    gat_prep_w2<<<(16 * 128 + T - 1) / T, T, 0, stream>>>(W2, a_s2, a_d2, wt2);
    gat_conv_f16<<<((size_t)N * 128 + T - 1) / T, T, 0, stream>>>(x, xh, N * 128);
    gat_init<<<((size_t)N * 128 + T - 1) / T, T, 0, stream>>>(out1, b1, out2, b2, m1u, den1, m2u, den2, N);

    // layer 1
    gat_gemm1<<<(N + 15) / 16, 288, 0, stream>>>(xh, wt1, h1, as1, ad1, N);
    gat_edge1a<<<(NE2 + T - 1) / T, T, 0, stream>>>(ei, (const float4*)as1, (const float4*)ad1,
                                                    (float4*)eb1, m1u, E, NE2);
    gat_edge1b<<<(NE2 + T - 1) / T, T, 0, stream>>>(ei, (float4*)eb1, m1u, den1, E, NE2);
    gat_edge1c<<<(NE2 + 7) / 8, T, 0, stream>>>(ei, (const float4*)eb1, (const float4*)den1,
                                                h1, out1, E, NE2);
    gat_elu1<<<((size_t)N * 128 + T - 1) / T, T, 0, stream>>>(out1, z1h, N * 128);

    // layer 2
    gat_gemm2<<<((N + 15) / 16 + 7) / 8, T, 0, stream>>>(z1h, wt2, h2, as2, ad2, N);
    gat_edge2a<<<(NE2 + T - 1) / T, T, 0, stream>>>(ei, as2, ad2, eb2, m2u, E, NE2);
    gat_edge2b<<<(NE2 + T - 1) / T, T, 0, stream>>>(ei, eb2, m2u, den2, E, NE2);
    gat_edge2c<<<((size_t)NE2 * 8 + T - 1) / T, T, 0, stream>>>(ei, eb2, den2, h2, out2, E, NE2);

    // readout
    gat_final<<<(N + T - 1) / T, T, 0, stream>>>(out2, Wlin, blin, out, N);
}
